// TopologyGuidedLoss_283467841734
// MI455X (gfx1250) — matrix operand order repaired and checked by reading, not compiled, not run on hardware
//
#include <hip/hip_runtime.h>
#include <hip/hip_bf16.h>
#include <stdint.h>

// ---------------------------------------------------------------- constants
static constexpr int B  = 16384;
static constexpr int C  = 250;
static constexpr int CP = 256;     // padded classes
static constexpr int D  = 2048;
#define LAMBDA_TOPO   0.1f
#define LAMBDA_MARGIN 0.05f
#define MARGIN_F      1.2f
#define TOPO_THRESH   0.3f
#define EPSF          1e-12f

// scal[] slots
// 0=CE 1=TMAX 2=PMAX(bits) 3=TOPO_SUM 4=NUPPER 5=MARGIN_SUM 6=NPAIRS 7=NPRESENT

typedef __attribute__((ext_vector_type(16))) __bf16 v16bf;
typedef __attribute__((ext_vector_type(8)))  float  v8f;
typedef __attribute__((ext_vector_type(2)))  float  v2f;

union FragB16 { v16bf v; uint32_t u[8]; };

__device__ __forceinline__ uint16_t f2bf(float f) {  // RNE f32 -> bf16 bits
    uint32_t u = __float_as_uint(f);
    uint32_t r = u + 0x7FFFu + ((u >> 16) & 1u);
    return (uint16_t)(r >> 16);
}

// ---------------------------------------------------------------- utilities
__global__ void zero_kernel(float* p, int n) {
    int i = blockIdx.x * 256 + threadIdx.x;
    if (i < n) p[i] = 0.0f;
}

__global__ void count_kernel(const int* __restrict__ labels, int* __restrict__ counts) {
    int i = blockIdx.x * 256 + threadIdx.x;
    if (i < B) atomicAdd(&counts[labels[i]], 1);
}

// ---------------------------------------------------------------- cross entropy
__global__ __launch_bounds__(256) void ce_kernel(const float* __restrict__ logits,
                                                 const int* __restrict__ labels,
                                                 float* __restrict__ scal) {
    int wave = threadIdx.x >> 5, lane = threadIdx.x & 31;
    int r = blockIdx.x * 8 + wave;
    if (r >= B) return;
    const float* row = logits + (size_t)r * C;
    float m = -3.0e38f;
    for (int c = lane; c < C; c += 32) m = fmaxf(m, row[c]);
    for (int o = 16; o > 0; o >>= 1) m = fmaxf(m, __shfl_xor(m, o, 32));
    float s = 0.0f;
    for (int c = lane; c < C; c += 32) s += __expf(row[c] - m);
    for (int o = 16; o > 0; o >>= 1) s += __shfl_xor(s, o, 32);
    if (lane == 0) {
        float lp = row[labels[r]] - m - __logf(s);
        atomicAdd(&scal[0], -lp * (1.0f / (float)B));
    }
}

// ---------------------------------------------------------------- max of topo matrix
__global__ void tmax_kernel(const float* __restrict__ M, float* __restrict__ scal) {
    __shared__ float red[1024];
    float m = 0.0f;
    for (int i = threadIdx.x; i < C * C; i += 1024) m = fmaxf(m, M[i]);
    red[threadIdx.x] = m;
    __syncthreads();
    for (int off = 512; off > 0; off >>= 1) {
        if ((int)threadIdx.x < off) red[threadIdx.x] = fmaxf(red[threadIdx.x], red[threadIdx.x + off]);
        __syncthreads();
    }
    if (threadIdx.x == 0) scal[1] = red[0];
}

// ---------------------------------------------------------------- centroid accumulation (+ row norms)
// grid = 512 blocks: (rowChunk = bx>>5 of 1024 rows) x (colChunk = bx&31 of 64 dims)
__global__ __launch_bounds__(256) void cent_accum_kernel(const float* __restrict__ emb,
                                                         const int* __restrict__ labels,
                                                         float* __restrict__ csum,
                                                         float* __restrict__ norm2) {
    __shared__ float lsum[C * 64];   // 62.5 KB
    int tid = threadIdx.x, wave = tid >> 5, lane = tid & 31;
    int rowStart = (blockIdx.x >> 5) * 1024;
    int colBase  = (blockIdx.x & 31) * 64;
    for (int i = tid; i < C * 64; i += 256) lsum[i] = 0.0f;
    __syncthreads();
    for (int r = rowStart + wave; r < rowStart + 1024; r += 8) {
        int lab = labels[r];
        float2 v = *(const float2*)(emb + (size_t)r * D + colBase + lane * 2);
        float ss = v.x * v.x + v.y * v.y;
        for (int o = 16; o > 0; o >>= 1) ss += __shfl_xor(ss, o, 32);
        if (lane == 0) atomicAdd(&norm2[r], ss);
        atomicAdd(&lsum[lab * 64 + lane * 2 + 0], v.x);
        atomicAdd(&lsum[lab * 64 + lane * 2 + 1], v.y);
    }
    __syncthreads();
    for (int i = tid; i < C * 64; i += 256) {
        float v = lsum[i];
        if (v != 0.0f) {
            int cls = i >> 6, col = i & 63;
            atomicAdd(&csum[(size_t)cls * D + colBase + col], v);
        }
    }
}

// ---------------------------------------------------------------- finalize centroids
__global__ __launch_bounds__(256) void cent_final_kernel(const float* __restrict__ csum,
                                                         const int* __restrict__ counts,
                                                         float* __restrict__ cent,
                                                         unsigned short* __restrict__ centn,
                                                         float* __restrict__ c2,
                                                         float* __restrict__ scal) {
    __shared__ float red[256];
    int i = blockIdx.x, tid = threadIdx.x;
    int cnt = counts[i];
    float denom = fmaxf((float)cnt, 1.0f);
    float vv[8], ss = 0.0f;
#pragma unroll
    for (int q = 0; q < 8; ++q) {
        int d = q * 256 + tid;
        float v = csum[(size_t)i * D + d] / denom;
        vv[q] = v;
        cent[(size_t)i * D + d] = v;
        ss += v * v;
    }
    red[tid] = ss;
    __syncthreads();
    for (int off = 128; off > 0; off >>= 1) {
        if (tid < off) red[tid] += red[tid + off];
        __syncthreads();
    }
    float n2  = red[0];
    float inv = 1.0f / fmaxf(sqrtf(n2), EPSF);
#pragma unroll
    for (int q = 0; q < 8; ++q) {
        int d = q * 256 + tid;
        centn[(size_t)i * D + d] = f2bf(vv[q] * inv);
    }
    if (tid == 0) {
        c2[i] = n2;
        if (i < C && cnt > 0) atomicAdd(&scal[7], 1.0f);
    }
}

// ---------------------------------------------------------------- margin weight table W[i][j] = mask / count[i]
__global__ void w_kernel(const float* __restrict__ M, const int* __restrict__ counts,
                         float* __restrict__ W, float* __restrict__ scal) {
    int i = blockIdx.x, j = threadIdx.x;
    float tmax = scal[1];
    bool mask = (i < C) && (j < C) && (i != j) && (counts[i] > 0) && (counts[j] > 0) &&
                (M[i * C + j] / tmax < TOPO_THRESH);
    W[i * CP + j] = mask ? 1.0f / (float)counts[i] : 0.0f;
    if (mask) atomicAdd(&scal[6], 1.0f);
}

// ---------------------------------------------------------------- centroid Gram -> pdist (f32 WMMA)
__global__ __launch_bounds__(256) void gram_pdist_kernel(const float* __restrict__ cent,
                                                         const float* __restrict__ c2,
                                                         const int* __restrict__ counts,
                                                         float* __restrict__ pdist,
                                                         float* __restrict__ scal) {
    int wave = threadIdx.x >> 5, lane = threadIdx.x & 31, lh = lane & 15, hi = lane >> 4;
    int tile = blockIdx.x * 8 + wave;        // 256 tiles of 16x16
    int ti = tile >> 4, tj = tile & 15;
    v8f acc = {};
#if __has_builtin(__builtin_amdgcn_wmma_f32_16x16x4_f32)
    {
        const float* pa = cent + (size_t)(ti * 16 + lh) * D;  // A: M=lh, K pair at hi*2
        const float* pb = cent + (size_t)(tj * 16 + lh) * D;  // B = cent^T: N=lh
        for (int k = 0; k < D; k += 4) {
            float2 fa = *(const float2*)(pa + k + hi * 2);
            float2 fb = *(const float2*)(pb + k + hi * 2);
            v2f a; a[0] = fa.x; a[1] = fa.y;
            v2f b; b[0] = fb.x; b[1] = fb.y;
            acc = __builtin_amdgcn_wmma_f32_16x16x4_f32(false, a, false, b, (short)0, acc,
                                                        false, false);
        }
    }
#else
#pragma unroll
    for (int r = 0; r < 8; ++r) {
        int i = ti * 16 + hi * 8 + r, j = tj * 16 + lh;
        const float* pi = cent + (size_t)i * D;
        const float* pj = cent + (size_t)j * D;
        float dsum = 0.0f;
        for (int k = 0; k < D; ++k) dsum += pi[k] * pj[k];
        acc[r] = dsum;
    }
#endif
#pragma unroll
    for (int r = 0; r < 8; ++r) {
        int i = ti * 16 + hi * 8 + r;        // D layout: VGPR r -> M = r + 8*hi
        int j = tj * 16 + lh;
        float g  = acc[r];
        float pd = sqrtf(fmaxf(c2[i] + c2[j] - 2.0f * g, 0.0f) + EPSF);
        pdist[i * CP + j] = pd;
        if (i < C && j < C && counts[i] > 0 && counts[j] > 0)
            atomicMax((int*)&scal[2], __float_as_int(pd));   // pd >= 0: bit-compare valid
    }
}

// ---------------------------------------------------------------- topo loss reduction
__global__ __launch_bounds__(256) void topo_loss_kernel(const float* __restrict__ pdist,
                                                        const float* __restrict__ M,
                                                        const int* __restrict__ counts,
                                                        float* __restrict__ scal) {
    __shared__ float rs[256], rc[256];
    float pmax = __int_as_float(((const int*)scal)[2]);
    float tmax = scal[1];
    float s = 0.0f, c = 0.0f;
    for (int idx = blockIdx.x * 256 + threadIdx.x; idx < C * C; idx += gridDim.x * 256) {
        int i = idx / C, j = idx - i * C;
        if (j > i && counts[i] > 0 && counts[j] > 0) {
            float d = pdist[i * CP + j] / pmax;
            float e = d - M[idx] / tmax;
            s += e * e;
            c += 1.0f;
        }
    }
    rs[threadIdx.x] = s; rc[threadIdx.x] = c;
    __syncthreads();
    for (int off = 128; off > 0; off >>= 1) {
        if ((int)threadIdx.x < off) {
            rs[threadIdx.x] += rs[threadIdx.x + off];
            rc[threadIdx.x] += rc[threadIdx.x + off];
        }
        __syncthreads();
    }
    if (threadIdx.x == 0) { atomicAdd(&scal[3], rs[0]); atomicAdd(&scal[4], rc[0]); }
}

// ---------------------------------------------------------------- big GEMM: emb_n x cent_n^T (bf16 WMMA), fused hinge/segment-sum
// grid = B/128 blocks, 256 threads = 8 waves; block tile 128 rows x 256 cols; K step 32
__global__ __launch_bounds__(256) void margin_gemm_kernel(const float* __restrict__ emb,
                                                          const int* __restrict__ labels,
                                                          const unsigned short* __restrict__ centn,
                                                          const float* __restrict__ norm2,
                                                          const float* __restrict__ W,
                                                          float* __restrict__ scal) {
    __shared__ unsigned short ldsA[128 * 32];   // bf16 A tile   (8 KB)
    __shared__ unsigned short ldsB[CP * 32];    // bf16 B tile  (16 KB)
    __shared__ float red[256];

    const int tid = threadIdx.x;
    const int wave = tid >> 5, lane = tid & 31, lh = lane & 15, hi = lane >> 4;
    const int rowBase = blockIdx.x * 128;

    v8f zero = {};
    v8f acc[16];
#pragma unroll
    for (int n = 0; n < 16; ++n) acc[n] = zero;

    for (int k0 = 0; k0 < D; k0 += 32) {
        // stage B: 256 cols x 32 K bf16, pure copy -> CDNA5 async global->LDS DMA.
        // INST_OFFSET applies to both LDS and global sides (ISA 08 §4.4), so one
        // address pair covers the 64-byte slice with 4 x b128 at offsets 0/16/32/48.
        {
            const unsigned short* src = centn + (size_t)tid * D + k0;
            unsigned long long ga = (unsigned long long)(uintptr_t)src;
            unsigned ldsOff = (unsigned)(uintptr_t)(&ldsB[tid * 32]);  // generic low 32 = LDS addr
            asm volatile("global_load_async_to_lds_b128 %0, %1, off"
                         :: "v"(ldsOff), "v"(ga) : "memory");
            asm volatile("global_load_async_to_lds_b128 %0, %1, off offset:16"
                         :: "v"(ldsOff), "v"(ga) : "memory");
            asm volatile("global_load_async_to_lds_b128 %0, %1, off offset:32"
                         :: "v"(ldsOff), "v"(ga) : "memory");
            asm volatile("global_load_async_to_lds_b128 %0, %1, off offset:48"
                         :: "v"(ldsOff), "v"(ga) : "memory");
        }
        // stage A: 128 rows x 32 K, f32 -> bf16 conversion into LDS (overlaps async B)
        {
            int row = tid >> 1;
            int kk  = (tid & 1) * 16;
            const float* src = emb + (size_t)(rowBase + row) * D + k0 + kk;
            unsigned short* dst = &ldsA[row * 32 + kk];
#pragma unroll
            for (int i = 0; i < 16; i += 4) {
                float4 f = *(const float4*)(src + i);
                dst[i + 0] = f2bf(f.x); dst[i + 1] = f2bf(f.y);
                dst[i + 2] = f2bf(f.z); dst[i + 3] = f2bf(f.w);
            }
            if (k0 + 32 < D) __builtin_prefetch(src + 32, 0, 0);
        }
        // wave's async B copies must land in LDS before the workgroup barrier
        asm volatile("s_wait_asynccnt 0x0" ::: "memory");
        __syncthreads();

        // A fragment: 16x32 bf16; lane half selects K 0..15 / 16..31 contiguous
        FragB16 a;
        {
            int arow = wave * 16 + lh;
            const uint32_t* p0 = (const uint32_t*)&ldsA[arow * 32 + hi * 16];
            const uint32_t* p1 = (const uint32_t*)&ldsA[arow * 32 + hi * 16 + 8];
            a.u[0] = p0[0]; a.u[1] = p0[1]; a.u[2] = p0[2]; a.u[3] = p0[3];
            a.u[4] = p1[0]; a.u[5] = p1[1]; a.u[6] = p1[2]; a.u[7] = p1[3];
        }
#pragma unroll
        for (int n = 0; n < 16; ++n) {
            FragB16 b;   // B fragment: 32x16; lane half selects K 0..15 / 16..31 contiguous
            const uint32_t* p = (const uint32_t*)&ldsB[(n * 16 + lh) * 32 + hi * 16];
#pragma unroll
            for (int v = 0; v < 8; ++v) b.u[v] = p[v];
            acc[n] = __builtin_amdgcn_wmma_f32_16x16x32_bf16(false, a.v, false, b.v, (short)0,
                                                             acc[n], false, false);
        }
        __syncthreads();
    }

    // epilogue: dist = sqrt(max(2 - 2*dot,0)); hinge * W[label][col]; block-reduce
    float invn[8]; int labr[8];
#pragma unroll
    for (int r = 0; r < 8; ++r) {
        int row = rowBase + wave * 16 + hi * 8 + r;   // D layout: M = r + 8*hi
        invn[r] = 1.0f / fmaxf(sqrtf(norm2[row]), EPSF);
        labr[r] = labels[row];
    }
    float s = 0.0f;
#pragma unroll
    for (int n = 0; n < 16; ++n) {
        int col = n * 16 + lh;
#pragma unroll
        for (int r = 0; r < 8; ++r) {
            float dot  = acc[n][r] * invn[r];
            float dist = sqrtf(fmaxf(2.0f - 2.0f * dot, 0.0f));
            float h    = fmaxf(MARGIN_F - dist, 0.0f);
            s += h * W[labr[r] * CP + col];
        }
    }
    red[tid] = s;
    __syncthreads();
    for (int off = 128; off > 0; off >>= 1) {
        if (tid < off) red[tid] += red[tid + off];
        __syncthreads();
    }
    if (tid == 0) atomicAdd(&scal[5], red[0]);
}

// ---------------------------------------------------------------- finalize
__global__ void finalize_kernel(const float* __restrict__ scal, float* __restrict__ out) {
    if (threadIdx.x == 0 && blockIdx.x == 0) {
        float ce    = scal[0];
        bool  gate  = scal[7] >= 2.0f;
        float topo  = scal[3] / fmaxf(scal[4], 1.0f);
        topo        = gate ? topo : 0.0f;
        float np    = scal[6];
        float marg  = scal[5] / fmaxf(np, 1.0f);
        marg        = (gate && np > 0.0f) ? marg : 0.0f;
        out[0] = ce + LAMBDA_TOPO * topo + LAMBDA_MARGIN * marg;
        out[1] = ce;
        out[2] = topo;
        out[3] = marg;
    }
}

// ---------------------------------------------------------------- launch
extern "C" void kernel_launch(void* const* d_in, const int* in_sizes, int n_in,
                              void* d_out, int out_size, void* d_ws, size_t ws_size,
                              hipStream_t stream) {
    (void)in_sizes; (void)n_in; (void)out_size; (void)ws_size;
    const float* logits = (const float*)d_in[0];
    const int*   labels = (const int*)d_in[1];
    const float* emb    = (const float*)d_in[2];
    const float* M      = (const float*)d_in[3];
    float* out = (float*)d_out;

    char* ws = (char*)d_ws;
    float*          scal   = (float*)(ws + 0);          // 64 f
    int*            counts = (int*)(ws + 1024);         // 256 i
    float*          c2     = (float*)(ws + 2048);       // 256 f
    float*          W      = (float*)(ws + 4096);       // 256*256 f  (256 KB)
    float*          norm2  = (float*)(ws + 266240);     // 16384 f    (64 KB)
    float*          pdist  = (float*)(ws + 331776);     // 256*256 f  (256 KB)
    float*          csum   = (float*)(ws + 593920);     // 256*2048 f (2 MB)
    float*          cent   = (float*)(ws + 2691072);    // 256*2048 f (2 MB)
    unsigned short* centn  = (unsigned short*)(ws + 4788224);  // 256*2048 bf16 (1 MB)

    zero_kernel<<<4, 256, 0, stream>>>((float*)ws, 1024);          // scal+counts+c2
    zero_kernel<<<64, 256, 0, stream>>>(norm2, B);
    zero_kernel<<<2048, 256, 0, stream>>>(csum, CP * D);

    count_kernel<<<B / 256, 256, 0, stream>>>(labels, counts);
    ce_kernel<<<B / 8, 256, 0, stream>>>(logits, labels, scal);
    tmax_kernel<<<1, 1024, 0, stream>>>(M, scal);
    cent_accum_kernel<<<512, 256, 0, stream>>>(emb, labels, csum, norm2);
    cent_final_kernel<<<CP, 256, 0, stream>>>(csum, counts, cent, centn, c2, scal);
    w_kernel<<<CP, 256, 0, stream>>>(M, counts, W, scal);
    gram_pdist_kernel<<<32, 256, 0, stream>>>(cent, c2, counts, pdist, scal);
    topo_loss_kernel<<<64, 256, 0, stream>>>(pdist, M, counts, scal);
    margin_gemm_kernel<<<B / 128, 256, 0, stream>>>(emb, labels, centn, norm2, W, scal);
    finalize_kernel<<<1, 32, 0, stream>>>(scal, out);
}
